// WaveletScattering2D_56573309223963
// MI455X (gfx1250) — matrix operand order repaired and checked by reading, not compiled, not run on hardware
//
#include <hip/hip_runtime.h>
#include <math.h>

// ---------------------------------------------------------------------------
// WaveletScattering2D for MI455X (gfx1250), wave32 + WMMA f16 implicit GEMM.
//   x:      [8,3,512,512] f32
//   w_real: [32,1,32,32]  f32
//   w_imag: [32,1,32,32]  f32
//   out:    S0 [8,1,32,32] ++ S1 [8,32,32,32]  (f32, flat concat)
//
// Conv as implicit GEMM per (batch, 16x16 spatial tile):
//   M = 256 positions, N = 64 filters, K = 1024 (32x32 taps), f16 x f16 -> f32.
// B-fragments staged global->LDS with GLOBAL_LOAD_ASYNC_TO_LDS_B128
// (ASYNCcnt), double-buffered, one barrier per K-step.
// ---------------------------------------------------------------------------

typedef __attribute__((ext_vector_type(16))) _Float16 v16h;
typedef __attribute__((ext_vector_type(8)))  float    v8f;
typedef int v4i_vs __attribute__((vector_size(16)));   // matches builtin param

#define BATCH 8
#define HW    512
#define PAD   16
#define PW    544           // 512 + 2*16 padded gray
#define FS    32            // filter size
#define PATCH_ROWS 47       // 16 + 31
#define PATCH_STRIDE 48
#define BSTAGE_DW (4 * 32 * 8)   // 4 fragments x 32 lanes x 8 dwords = 1024 dw

#if __has_builtin(__builtin_amdgcn_global_load_async_to_lds_b128)
#define HAVE_ASYNC_LDS 1
#endif

union Frag16 {              // one 16x16x32 f16 WMMA operand fragment (8 VGPRs)
    v16h h;
    _Float16 e[16];
    unsigned int u[8];
};

// ---------------------------------------------------------------------------
// Kernel 1: gray = mean(channels), f16, zero-padded to 544x544 per batch.
// ---------------------------------------------------------------------------
__global__ void pad_gray_kernel(const float* __restrict__ x,
                                _Float16* __restrict__ pg) {
    int i = blockIdx.x * 256 + threadIdx.x;
    const int total = BATCH * PW * PW;
    if (i >= total) return;
    int c = i % PW;
    int t = i / PW;
    int r = t % PW;
    int b = t / PW;
    float v = 0.0f;
    int iy = r - PAD, ix = c - PAD;
    if (iy >= 0 && iy < HW && ix >= 0 && ix < HW) {
        size_t base = ((size_t)b * 3) * (size_t)(HW * HW) + (size_t)iy * HW + ix;
        v = (x[base] + x[base + (size_t)(HW * HW)] + x[base + 2 * (size_t)(HW * HW)])
            * (1.0f / 3.0f);
    }
    pg[i] = (_Float16)v;
}

// ---------------------------------------------------------------------------
// Kernel 2: pre-swizzle filters into WMMA B-fragment order.
// B fragment, contiguous k order used here (K=32 x N=16, f16): lanes 0-15 hold K=0..15 (VGPR v -> K=2v,2v+1),
// lanes 16-31 hold K=16..31 (VGPR v -> K=16+2v,17+2v); N = lane&15.
// Storage: fb[((ntile*32 + kstep)*32 + lane)*8 + v].
// ---------------------------------------------------------------------------
__global__ void pack_filters_kernel(const float* __restrict__ wr,
                                    const float* __restrict__ wi,
                                    unsigned int* __restrict__ fb) {
    int i = blockIdx.x * 256 + threadIdx.x;   // 4*32*32*8 = 32768
    if (i >= 4 * 32 * 32 * 8) return;
    int v    = i & 7;
    int lane = (i >> 3) & 31;
    int ks   = (i >> 8) & 31;                 // kernel row (K-step)
    int nt   = (i >> 13) & 3;                 // N tile: filters nt*16..nt*16+15
    int lanehi = lane >> 4;
    int n  = lane & 15;
    int f  = nt * 16 + n;                     // 0..63 (concat real,imag)
    int kx0 = 2 * v + (lanehi ? 16 : 0);      // column within filter row
    const float* w = (nt < 2) ? wr : wi;
    int fi = f & 31;
    float a0 = w[(fi * FS + ks) * FS + kx0];
    float a1 = w[(fi * FS + ks) * FS + kx0 + 1];
    union { _Float16 h[2]; unsigned int u; } pk;
    pk.h[0] = (_Float16)a0;
    pk.h[1] = (_Float16)a1;
    fb[i] = pk.u;
}

// ---------------------------------------------------------------------------
// Kernel 3: S0 = avg-pool(gray) with 16x16 windows.  One block per output.
// ---------------------------------------------------------------------------
__global__ void s0_kernel(const float* __restrict__ x, float* __restrict__ out) {
    __shared__ float s;
    int px = blockIdx.x, py = blockIdx.y, b = blockIdx.z;
    int tid = threadIdx.x;
    if (tid == 0) s = 0.0f;
    __syncthreads();
    int iy = py * 16 + (tid >> 4);
    int ix = px * 16 + (tid & 15);
    size_t base = (size_t)b * 3 * (size_t)(HW * HW) + (size_t)iy * HW + ix;
    float v = x[base] + x[base + (size_t)(HW * HW)] + x[base + 2 * (size_t)(HW * HW)];
    atomicAdd(&s, v);
    __syncthreads();
    if (tid == 0)
        out[((size_t)b * 32 + py) * 32 + px] = s * (1.0f / 768.0f);  // /3 /256
}

// ---------------------------------------------------------------------------
// Stage the 4 KB of B-fragments for K-step `ks` into LDS (one b128 / thread).
// LDS layout matches the per-lane read pattern: dst[(nt*32+lane)*8 + v].
// ---------------------------------------------------------------------------
__device__ __forceinline__ void stage_b(const unsigned int* __restrict__ fb,
                                        unsigned int* dst, int ks, int tid) {
    int nt   = tid >> 6;
    int rem  = tid & 63;
    int lane = rem >> 1;
    int half = rem & 1;
    const unsigned int* g = fb + (((nt * 32 + ks) * 32 + lane) << 3) + (half << 2);
    unsigned int*       l = dst + (((nt * 32 + lane) << 3) + (half << 2));
#ifdef HAVE_ASYNC_LDS
    __builtin_amdgcn_global_load_async_to_lds_b128(
        (__attribute__((address_space(1))) v4i_vs*)g,
        (__attribute__((address_space(3))) v4i_vs*)l, 0, 0);
#else
    *(uint4*)l = *(const uint4*)g;
#endif
}

__device__ __forceinline__ void wait_async_lds() {
#ifdef HAVE_ASYNC_LDS
#if __has_builtin(__builtin_amdgcn_s_wait_asynccnt)
    __builtin_amdgcn_s_wait_asynccnt(0);
#else
    asm volatile("s_wait_asynccnt 0" ::: "memory");
#endif
#endif
}

// ---------------------------------------------------------------------------
// Kernel 4: fused implicit-GEMM conv + |.| + 16x16 avg-pool -> S1.
// Block = 256 threads = 8 waves; computes one (batch, 16x16 spatial tile).
// Wave w owns M-tiles {2w,2w+1} x all 4 N-tiles -> 8 v_wmma per K-step.
// ---------------------------------------------------------------------------
__global__ __launch_bounds__(256) void scat_conv_kernel(
    const _Float16* __restrict__ pg,
    const unsigned int* __restrict__ fb,
    float* __restrict__ s1out) {

    __shared__ __align__(16) _Float16 patch[PATCH_ROWS * PATCH_STRIDE]; // 4512 B
    __shared__ __align__(16) unsigned int bstage[2][BSTAGE_DW];         // 8 KB
    __shared__ float pool[64];

    int tx = blockIdx.x, ty = blockIdx.y, b = blockIdx.z;
    int tid  = threadIdx.x;
    int lane = tid & 31;
    int wv   = tid >> 5;
    int lanehi = lane >> 4;
    int n16    = lane & 15;

    // ---- stage input patch (47x47 f16, stride 48) into LDS ----
    int Y0 = ty * 16, X0 = tx * 16;
    const _Float16* src = pg + (size_t)b * PW * PW;
    for (int i = tid; i < PATCH_ROWS * PATCH_STRIDE; i += 256) {
        int r = i / PATCH_STRIDE, c = i % PATCH_STRIDE;
        _Float16 val = (_Float16)0.0f;
        if (c < 47) val = src[(Y0 + r) * PW + X0 + c];
        patch[i] = val;
    }
    if (tid < 64) pool[tid] = 0.0f;

    // prime B double-buffer with K-step 0
    stage_b(fb, bstage[0], 0, tid);
    wait_async_lds();
    __syncthreads();

    v8f zero = {0.f, 0.f, 0.f, 0.f, 0.f, 0.f, 0.f, 0.f};
    v8f acc[2][4];
#pragma unroll
    for (int mi = 0; mi < 2; ++mi)
#pragma unroll
        for (int nt = 0; nt < 4; ++nt) acc[mi][nt] = zero;

    // ---- K loop: one kernel row per step (K = ks*32 + kx) ----
#pragma unroll 1
    for (int ks = 0; ks < 32; ++ks) {
        const unsigned int* bcur = bstage[ks & 1];
        // kick off async copy of next step's fragments into the other buffer
        if (ks < 31) {
            stage_b(fb, bstage[(ks + 1) & 1], ks + 1, tid);
            if (ks < 30)
                __builtin_prefetch(fb + ((((ks + 2) * 32) + lane) << 3), 0, 1);
        }

        // B fragments from LDS: 4 N-tiles, two ds_load_b128 each.
        Frag16 bf[4];
#pragma unroll
        for (int nt = 0; nt < 4; ++nt) {
            const uint4* p = (const uint4*)(bcur + ((nt * 32 + lane) << 3));
            uint4 q0 = p[0], q1 = p[1];
            bf[nt].u[0] = q0.x; bf[nt].u[1] = q0.y;
            bf[nt].u[2] = q0.z; bf[nt].u[3] = q0.w;
            bf[nt].u[4] = q1.x; bf[nt].u[5] = q1.y;
            bf[nt].u[6] = q1.z; bf[nt].u[7] = q1.w;
        }

        // A fragments from LDS in the same contiguous k order as B:
        // lanes 0-15: VGPR v -> K = 2v; lanes 16-31: +16.
        Frag16 af[2];
#pragma unroll
        for (int mi = 0; mi < 2; ++mi) {
            int p  = (wv * 2 + mi) * 16 + n16;   // position 0..255 in tile
            int oy = p >> 4, ox = p & 15;
            int rowbase = (oy + ks) * PATCH_STRIDE + ox;
#pragma unroll
            for (int v = 0; v < 8; ++v) {
                int kx0 = 2 * v + (lanehi ? 16 : 0);
                af[mi].e[2 * v]     = patch[rowbase + kx0];
                af[mi].e[2 * v + 1] = patch[rowbase + kx0 + 1];
            }
        }

#pragma unroll
        for (int mi = 0; mi < 2; ++mi)
#pragma unroll
            for (int nt = 0; nt < 4; ++nt)
                acc[mi][nt] = __builtin_amdgcn_wmma_f32_16x16x32_f16(
                    false, af[mi].h, false, bf[nt].h,
                    (short)0, acc[mi][nt], false, false);

        // ensure this wave's async copy landed, then block-wide barrier so
        // next iteration can read the freshly staged buffer (and safely
        // overwrite the one all waves just finished reading).
        wait_async_lds();
        __syncthreads();
    }

    // ---- epilogue: |real + i*imag| then 16x16 average pool ----
    // N-tile 0/1 = real filters 0..31; N-tile 2/3 = matching imag filters.
    // D layout: N = lane&15; lanes 0-15 hold M=g, lanes 16-31 hold M=8+g.
#pragma unroll
    for (int nt = 0; nt < 2; ++nt) {
        float s = 0.0f;
#pragma unroll
        for (int mi = 0; mi < 2; ++mi)
#pragma unroll
            for (int g = 0; g < 8; ++g) {
                float re = acc[mi][nt][g];
                float im = acc[mi][nt + 2][g];
                s += sqrtf(re * re + im * im + 1e-8f);
            }
        atomicAdd(&pool[nt * 16 + n16], s);   // ds_add_f32
    }
    __syncthreads();
    if (tid < 32) {
        // S1[b][f][ty][tx]
        s1out[(((size_t)b * 32 + tid) * 32 + ty) * 32 + tx] =
            pool[tid] * (1.0f / 256.0f);
    }
}

// ---------------------------------------------------------------------------
extern "C" void kernel_launch(void* const* d_in, const int* in_sizes, int n_in,
                              void* d_out, int out_size, void* d_ws, size_t ws_size,
                              hipStream_t stream) {
    (void)in_sizes; (void)n_in; (void)out_size; (void)ws_size;
    const float* x  = (const float*)d_in[0];
    const float* wr = (const float*)d_in[1];
    const float* wi = (const float*)d_in[2];
    float* out = (float*)d_out;

    // workspace layout: padded gray f16, then packed filter fragments.
    size_t pg_bytes = (size_t)BATCH * PW * PW * sizeof(_Float16);  // ~4.73 MB
    size_t off = (pg_bytes + 255) & ~(size_t)255;
    _Float16*     pg = (_Float16*)d_ws;
    unsigned int* fb = (unsigned int*)((char*)d_ws + off);         // 128 KB

    {
        int total = BATCH * PW * PW;
        hipLaunchKernelGGL(pad_gray_kernel, dim3((total + 255) / 256), dim3(256),
                           0, stream, x, pg);
    }
    hipLaunchKernelGGL(pack_filters_kernel, dim3(32768 / 256), dim3(256),
                       0, stream, wr, wi, fb);
    hipLaunchKernelGGL(s0_kernel, dim3(32, 32, BATCH), dim3(256),
                       0, stream, x, out);
    hipLaunchKernelGGL(scat_conv_kernel, dim3(32, 32, BATCH), dim3(256),
                       0, stream, pg, fb, out + BATCH * 32 * 32);
}
